// HGTMessage_30562987278728
// MI455X (gfx1250) — matrix operand order repaired and checked by reading, not compiled, not run on hardware
//
#include <hip/hip_runtime.h>

// ---------------- CDNA5 WMMA types ----------------
typedef __attribute__((ext_vector_type(16))) __bf16 v16bf;
typedef __attribute__((ext_vector_type(8)))  float  v8f;

union FragB {
  v16bf v;
  uint4 q[2];
};

static __device__ __forceinline__ unsigned short f2bf(float x) {
  unsigned u = __builtin_bit_cast(unsigned, x);
  unsigned r = 0x7FFFu + ((u >> 16) & 1u);   // round-to-nearest-even
  return (unsigned short)((u + r) >> 16);
}
static __device__ __forceinline__ unsigned pack2(float a, float b) {
  return (unsigned)f2bf(a) | ((unsigned)f2bf(b) << 16);
}
static __device__ __forceinline__ v8f wmma_bf16(const FragB& a, const FragB& b, v8f c) {
  return __builtin_amdgcn_wmma_f32_16x16x32_bf16(false, a.v, false, b.v,
                                                 (short)0, c, false, false);
}

// LDS row pitch in bf16 elements (512 data + 8 pad -> 1040B/row, breaks bank conflicts)
#define AP 520

// ---------------- prep: T[t] = emb[t] @ lin_W^T + lin_b (240x512, tiny) ----------------
__global__ void hgt_rte_table(const float* __restrict__ emb,
                              const float* __restrict__ linW,
                              const float* __restrict__ linb,
                              float* __restrict__ T) {
  int t = blockIdx.x;
  const float* er = emb + t * 512;
  for (int o = threadIdx.x; o < 512; o += blockDim.x) {
    const float* wr = linW + o * 512;
    float acc = linb[o];
#pragma unroll 8
    for (int i = 0; i < 512; ++i) acc += er[i] * wr[i];
    T[t * 512 + o] = acc;
  }
}

// ---------------- prep: fp32 -> bf16 weight conversion ----------------
__global__ void hgt_cvt_bf16(const float* __restrict__ src,
                             unsigned short* __restrict__ dst, int n) {
  int i = blockIdx.x * blockDim.x + threadIdx.x;
  if (i < n) dst[i] = f2bf(src[i]);
}

// ---------------- binning prologue: key = tau*8 + etype (32 bins) ----------------
__global__ void hgt_zero32(int* __restrict__ p) {
  if (threadIdx.x < 32) p[threadIdx.x] = 0;
}
__global__ void hgt_count(const int* __restrict__ tau, const int* __restrict__ et,
                          int* __restrict__ counts, int E) {
  int i = blockIdx.x * blockDim.x + threadIdx.x;
  if (i < E) atomicAdd(&counts[tau[i] * 8 + et[i]], 1);
}
__global__ void hgt_scan32(const int* __restrict__ counts, int* __restrict__ cursor) {
  if (threadIdx.x == 0) {
    int acc = 0;
    for (int i = 0; i < 32; ++i) { cursor[i] = acc; acc += counts[i]; }
  }
}
__global__ void hgt_scatter(const int* __restrict__ tau, const int* __restrict__ et,
                            int* __restrict__ cursor, int* __restrict__ perm, int E) {
  int i = blockIdx.x * blockDim.x + threadIdx.x;
  if (i < E) {
    int k = tau[i] * 8 + et[i];
    int p = atomicAdd(&cursor[k], 1);
    perm[p] = i;
  }
}

// ---------------- main fused kernel: 64 (permuted) edges per block, 8 wave32s ----------------
__global__ __launch_bounds__(256)
void hgt_msg_main(const float* __restrict__ h_s,
                  const float* __restrict__ Q_t,
                  const int*   __restrict__ etype,
                  const int*   __restrict__ tau_s,
                  const int*   __restrict__ dt,
                  const int*   __restrict__ perm,
                  const float* __restrict__ T,    // [240,512] fp32
                  const unsigned short* __restrict__ Kw,  // bf16 [4,512,512]
                  const float* __restrict__ Kb,           // [4,512]
                  const unsigned short* __restrict__ Vw,  // bf16 [4,512,512]
                  const float* __restrict__ Vb,           // [4,512]
                  const unsigned short* __restrict__ Waw, // bf16 [8,64,64]
                  const unsigned short* __restrict__ Wmw, // bf16 [8,64,64]
                  const float* __restrict__ mu,           // [8,8]
                  float* __restrict__ out_att,            // [E,8]
                  float* __restrict__ out_M,              // [E,8,64]
                  int E) {
  extern __shared__ char smem[];
  unsigned short* sA = (unsigned short*)smem;                  // [64][AP] h_hat bf16
  unsigned short* sK = (unsigned short*)(smem + 64 * AP * 2);  // [64][AP] K_s bf16
  unsigned short* sV = (unsigned short*)(smem + 128 * AP * 2); // [64][AP] V_s bf16
  int* sTau  = (int*)(smem + 192 * AP * 2);
  int* sEt   = sTau + 64;
  int* sEdge = sEt + 64;

  const int tid  = threadIdx.x;
  const int lane = tid & 31;
  const int wave = tid >> 5;
  const int eb   = blockIdx.x * 64;

  for (int i = tid; i < 64; i += 256) {
    int pe = (eb + i < E) ? perm[eb + i] : -1;
    sEdge[i] = pe;
    sTau[i]  = (pe >= 0) ? tau_s[pe] : -1;
    sEt[i]   = (pe >= 0) ? etype[pe] : -1;
  }
  __syncthreads();

  // ---- stage h_hat = h_s + T[dt] into LDS as bf16 (4 threads per edge) ----
  {
    int el = tid >> 2, part = tid & 3;
    int pe = sEdge[el];
    uint2* dst = (uint2*)(sA + el * AP + part * 128);
    if (pe >= 0) {
      const float4* hp = (const float4*)(h_s + (size_t)pe * 512) + part * 32;
      const float4* tp = (const float4*)(T + (size_t)dt[pe] * 512) + part * 32;
#pragma unroll 4
      for (int j = 0; j < 32; ++j) {
        float4 a = hp[j], b = tp[j];
        uint2 p;
        p.x = pack2(a.x + b.x, a.y + b.y);
        p.y = pack2(a.z + b.z, a.w + b.w);
        dst[j] = p;
      }
    } else {
      uint2 z; z.x = 0u; z.y = 0u;
#pragma unroll 4
      for (int j = 0; j < 32; ++j) dst[j] = z;
    }
  }
  __syncthreads();

  const int  n16 = lane & 15;       // A row-in-tile / B,C column-in-tile
  const int  hi  = lane >> 4;       // lane half (K/M split per WMMA layout)
  const uint4 z4 = make_uint4(0u, 0u, 0u, 0u);

  // ================= Stage 1: K_s / V_s typed projections =================
  {
    const int mat = wave >> 2;      // 0 -> K path, 1 -> V path
    const int mt  = wave & 3;       // edge 16-tile
    const unsigned short* Wb = mat ? Vw : Kw;
    const float*          Bb = mat ? Vb : Kb;
    unsigned short*     sOut = mat ? sV : sK;

    const int arow  = mt * 16 + n16;
    const int myTau = sTau[arow];
    const unsigned short* aBase = sA + arow * AP;

    for (int nt = 0; nt < 32; ++nt) {
      v8f acc = {0.f, 0.f, 0.f, 0.f, 0.f, 0.f, 0.f, 0.f};
      const int ncol = nt * 16 + n16;
      if (nt < 31)  // stream-ahead the next N-tile's weight row
        __builtin_prefetch(Wb + ((size_t)(nt * 16 + 16 + n16)) * 512, 0, 3);
      for (int t = 0; t < 4; ++t) {
        // wave-uniform skip: after binning, almost every tile has one type
        if (__builtin_amdgcn_ballot_w32(myTau == t) == 0u) continue;
        const bool keep = (myTau == t);
        const unsigned short* wrow =
            Wb + ((size_t)t * 512 + ncol) * 512 + (hi ? 8 : 0);
#pragma unroll 4
        for (int ks = 0; ks < 16; ++ks) {
          FragB a, b;
          const int k0 = ks * 32 + (hi ? 8 : 0);
          uint4 a0 = *(const uint4*)(aBase + k0);
          uint4 a1 = *(const uint4*)(aBase + k0 + 16);
          a.q[0] = keep ? a0 : z4;
          a.q[1] = keep ? a1 : z4;
          const uint4* bp = (const uint4*)(wrow + ks * 32);
          b.q[0] = bp[0];
          b.q[1] = bp[2];
          acc = wmma_bf16(a, b, acc);
        }
      }
      // write C tile (+ typed bias) back to LDS as bf16
#pragma unroll
      for (int r = 0; r < 8; ++r) {
        int row = mt * 16 + r + hi * 8;
        int tt  = sTau[row];
        float bv = (tt >= 0) ? Bb[tt * 512 + ncol] : 0.f;
        sOut[row * AP + ncol] = f2bf(acc[r] + bv);
      }
    }
  }
  __syncthreads();

  // ============ Stage 2: per-head typed 64x64 transforms + attention ============
  {
    const int h = wave;  // 8 waves <-> 8 heads
    for (int m = 0; m < 4; ++m) {
      const int row2 = m * 16 + n16;
      const int myEt = sEt[row2];
      const unsigned short* vBase = sV + row2 * AP + h * 64;
      const unsigned short* kBase = sK + row2 * AP + h * 64;

      // ---- message: M = V_s @ Wmsg[etype]^T ----
      for (int nt = 0; nt < 4; ++nt) {
        v8f acc = {0.f, 0.f, 0.f, 0.f, 0.f, 0.f, 0.f, 0.f};
        const int ncol = nt * 16 + n16;
        for (int et = 0; et < 8; ++et) {
          if (__builtin_amdgcn_ballot_w32(myEt == et) == 0u) continue;
          const bool keep = (myEt == et);
          const unsigned short* wrow =
              Wmw + ((size_t)et * 64 + ncol) * 64 + (hi ? 8 : 0);
#pragma unroll
          for (int ks = 0; ks < 2; ++ks) {
            FragB a, b;
            const int k0 = ks * 32 + (hi ? 8 : 0);
            uint4 a0 = *(const uint4*)(vBase + k0);
            uint4 a1 = *(const uint4*)(vBase + k0 + 16);
            a.q[0] = keep ? a0 : z4;
            a.q[1] = keep ? a1 : z4;
            const uint4* bp = (const uint4*)(wrow + ks * 32);
            b.q[0] = bp[0];
            b.q[1] = bp[2];
            acc = wmma_bf16(a, b, acc);
          }
        }
#pragma unroll
        for (int r = 0; r < 8; ++r) {
          int pe = sEdge[m * 16 + r + hi * 8];
          if (pe >= 0) out_M[(size_t)pe * 512 + h * 64 + ncol] = acc[r];
        }
      }

      // ---- attention: (Q . (K_s @ Watt[etype]^T)) * mu / 8 ----
      float part[8];
#pragma unroll
      for (int r = 0; r < 8; ++r) part[r] = 0.f;

      for (int nt = 0; nt < 4; ++nt) {
        v8f acc = {0.f, 0.f, 0.f, 0.f, 0.f, 0.f, 0.f, 0.f};
        const int ncol = nt * 16 + n16;
        for (int et = 0; et < 8; ++et) {
          if (__builtin_amdgcn_ballot_w32(myEt == et) == 0u) continue;
          const bool keep = (myEt == et);
          const unsigned short* wrow =
              Waw + ((size_t)et * 64 + ncol) * 64 + (hi ? 8 : 0);
#pragma unroll
          for (int ks = 0; ks < 2; ++ks) {
            FragB a, b;
            const int k0 = ks * 32 + (hi ? 8 : 0);
            uint4 a0 = *(const uint4*)(kBase + k0);
            uint4 a1 = *(const uint4*)(kBase + k0 + 16);
            a.q[0] = keep ? a0 : z4;
            a.q[1] = keep ? a1 : z4;
            const uint4* bp = (const uint4*)(wrow + ks * 32);
            b.q[0] = bp[0];
            b.q[1] = bp[2];
            acc = wmma_bf16(a, b, acc);
          }
        }
#pragma unroll
        for (int r = 0; r < 8; ++r) {
          int pe = sEdge[m * 16 + r + hi * 8];
          float q = (pe >= 0) ? Q_t[(size_t)pe * 512 + h * 64 + ncol] : 0.f;
          part[r] += acc[r] * q;
        }
      }
      // reduce over the 16-lane N group (xor stays inside the half)
#pragma unroll
      for (int r = 0; r < 8; ++r) {
        float v = part[r];
        v += __shfl_xor(v, 1, 32);
        v += __shfl_xor(v, 2, 32);
        v += __shfl_xor(v, 4, 32);
        v += __shfl_xor(v, 8, 32);
        if (n16 == 0) {
          int row = m * 16 + r + hi * 8;
          int pe = sEdge[row];
          if (pe >= 0) {
            int et = sEt[row];
            out_att[(size_t)pe * 8 + h] = v * mu[et * 8 + h] * 0.125f;
          }
        }
      }
    }
  }
}

// ---------------- host side ----------------
extern "C" void kernel_launch(void* const* d_in, const int* in_sizes, int n_in,
                              void* d_out, int out_size, void* d_ws, size_t ws_size,
                              hipStream_t stream) {
  const float* h_s   = (const float*)d_in[0];
  const float* Q_t   = (const float*)d_in[1];
  const int*   etype = (const int*)d_in[2];
  const int*   tau   = (const int*)d_in[3];
  const int*   dtp   = (const int*)d_in[4];
  const float* emb   = (const float*)d_in[5];
  const float* linW  = (const float*)d_in[6];
  const float* linb  = (const float*)d_in[7];
  const float* K_W   = (const float*)d_in[8];
  const float* K_b   = (const float*)d_in[9];
  const float* V_W   = (const float*)d_in[10];
  const float* V_b   = (const float*)d_in[11];
  const float* Watt  = (const float*)d_in[12];
  const float* Wmsg  = (const float*)d_in[13];
  const float* mu    = (const float*)d_in[14];
  const int E = in_sizes[2];  // etype has E elements

  // workspace layout (bytes)
  char* ws = (char*)d_ws;
  float*          T    = (float*)ws;                                    // 240*512*4   = 491520
  unsigned short* Kw   = (unsigned short*)(ws + 491520);                // 4*512*512*2 = 2097152
  unsigned short* Vw   = (unsigned short*)(ws + 491520 + 2097152);
  unsigned short* Waw  = (unsigned short*)(ws + 491520 + 2 * 2097152);  // 8*64*64*2 = 65536
  unsigned short* Wmw  = (unsigned short*)(ws + 491520 + 2 * 2097152 + 65536);
  int*            cnts = (int*)(ws + 491520 + 2 * 2097152 + 2 * 65536); // 32 ints
  int*            curs = cnts + 32;                                     // 32 ints
  int*            perm = curs + 32;                                     // E ints

  hgt_rte_table<<<240, 256, 0, stream>>>(emb, linW, linb, T);
  const int nKW = 4 * 512 * 512;
  hgt_cvt_bf16<<<(nKW + 255) / 256, 256, 0, stream>>>(K_W, Kw, nKW);
  hgt_cvt_bf16<<<(nKW + 255) / 256, 256, 0, stream>>>(V_W, Vw, nKW);
  const int nSm = 8 * 64 * 64;
  hgt_cvt_bf16<<<(nSm + 255) / 256, 256, 0, stream>>>(Watt, Waw, nSm);
  hgt_cvt_bf16<<<(nSm + 255) / 256, 256, 0, stream>>>(Wmsg, Wmw, nSm);

  // bin edges by (tau, etype) so tiles are (nearly) single-typed
  const int nb = (E + 255) / 256;
  hgt_zero32<<<1, 32, 0, stream>>>(cnts);
  hgt_count<<<nb, 256, 0, stream>>>(tau, etype, cnts, E);
  hgt_scan32<<<1, 32, 0, stream>>>(cnts, curs);
  hgt_scatter<<<nb, 256, 0, stream>>>(tau, etype, curs, perm, E);

  float* out_att = (float*)d_out;
  float* out_M   = out_att + (size_t)E * 8;

  const int nblocks = (E + 63) / 64;
  const size_t shmem = (size_t)192 * AP * 2 + 768;  // 3 x [64][520] bf16 + type/edge arrays
  (void)hipFuncSetAttribute((const void*)hgt_msg_main,
                            hipFuncAttributeMaxDynamicSharedMemorySize, (int)shmem);
  hgt_msg_main<<<nblocks, 256, shmem, stream>>>(h_s, Q_t, etype, tau, dtp, perm, T,
                                                Kw, K_b, Vw, V_b, Waw, Wmw, mu,
                                                out_att, out_M, E);
}
